// MultiHeadCausalTensionLayer_60507499266064
// MI455X (gfx1250) — matrix operand order repaired and checked by reading, not compiled, not run on hardware
//
#include <hip/hip_runtime.h>

typedef __attribute__((ext_vector_type(16))) __bf16 v16bf;
typedef __attribute__((ext_vector_type(8)))  __bf16 v8bf;
typedef __attribute__((ext_vector_type(8)))  float  v8f;

#define DIMD 1024
#define TLEN 4096
#define BN   2
#define MTOT (BN * TLEN)   // 8192
#define NHEAD 16
#define HDIM 64
#define WIN 16

// ---------- bf16 helpers ----------
__device__ __forceinline__ unsigned short f2bf_u(float f) {
    unsigned int u = __float_as_uint(f);
    unsigned int r = u + 0x7FFFu + ((u >> 16) & 1u);   // round-to-nearest-even
    return (unsigned short)(r >> 16);
}
__device__ __forceinline__ __bf16 f2bf(float f) {
    unsigned short s = f2bf_u(f);
    return __builtin_bit_cast(__bf16, s);
}
__device__ __forceinline__ float bf2f(unsigned short s) {
    return __uint_as_float(((unsigned int)s) << 16);
}
__device__ __forceinline__ unsigned int pack2(float lo, float hi) {
    return (unsigned int)f2bf_u(lo) | ((unsigned int)f2bf_u(hi) << 16);
}

// ============================================================
// Kernel 1: qkv = x @ [wq;wkv]^T   (M=8192, N=3072, K=1024)
// block tile 128(M) x 64(N), K-step 32, 8 waves = 4x2, wave tile 32x32
// ============================================================
__global__ __launch_bounds__(256)
void k_gemm_qkv(const float* __restrict__ x, const float* __restrict__ wq,
                const float* __restrict__ wkv, unsigned short* __restrict__ qkv)
{
    __shared__ __bf16 As[128][32];   // 8 KB
    __shared__ __bf16 Bs[64][32];    // 4 KB

    const int tid  = threadIdx.x;
    const int lane = tid & 31;
    const int wv   = tid >> 5;
    const int wvM  = wv >> 1;        // 0..3
    const int wvN  = wv & 1;         // 0..1
    const int m0   = blockIdx.y * 128;
    const int n0   = blockIdx.x * 64;

    v8f acc[2][2];
    #pragma unroll
    for (int i = 0; i < 2; ++i)
        #pragma unroll
        for (int j = 0; j < 2; ++j)
            acc[i][j] = (v8f){0.f,0.f,0.f,0.f,0.f,0.f,0.f,0.f};

    const int aRow = tid >> 1;            // 0..127
    const int aCol = (tid & 1) * 16;      // 0 or 16
    const int bRow = tid >> 2;            // 0..63
    const int bCol = (tid & 3) * 8;       // 0,8,16,24

    const int nG = n0 + bRow;
    const float* wrow = (nG < DIMD) ? (wq + (size_t)nG * DIMD)
                                    : (wkv + (size_t)(nG - DIMD) * DIMD);
    const float* arowp = x + (size_t)(m0 + aRow) * DIMD + aCol;

    const int klo = (lane & 16) ? 16 : 0;  // A fragment K sub-offset
    const int kbB = (lane & 16) ? 16 : 0;  // B fragment K sub-offset
    const int lr  = lane & 15;

    for (int kk = 0; kk < DIMD; kk += 32) {
        __syncthreads();
        // --- stage A tile (fp32 -> bf16) ---
        const float* ap = arowp + kk;
        float4 a0 = *(const float4*)(ap + 0);
        float4 a1 = *(const float4*)(ap + 4);
        float4 a2 = *(const float4*)(ap + 8);
        float4 a3 = *(const float4*)(ap + 12);
        uint4 pa0, pa1;
        pa0.x = pack2(a0.x, a0.y); pa0.y = pack2(a0.z, a0.w);
        pa0.z = pack2(a1.x, a1.y); pa0.w = pack2(a1.z, a1.w);
        pa1.x = pack2(a2.x, a2.y); pa1.y = pack2(a2.z, a2.w);
        pa1.z = pack2(a3.x, a3.y); pa1.w = pack2(a3.z, a3.w);
        *(uint4*)&As[aRow][aCol]     = pa0;
        *(uint4*)&As[aRow][aCol + 8] = pa1;
        // --- stage B tile (fp32 -> bf16) ---
        const float* bp = wrow + kk + bCol;
        float4 b0 = *(const float4*)(bp + 0);
        float4 b1 = *(const float4*)(bp + 4);
        uint4 pb;
        pb.x = pack2(b0.x, b0.y); pb.y = pack2(b0.z, b0.w);
        pb.z = pack2(b1.x, b1.y); pb.w = pack2(b1.z, b1.w);
        *(uint4*)&Bs[bRow][bCol] = pb;
        __syncthreads();

        #pragma unroll
        for (int mi = 0; mi < 2; ++mi) {
            const __bf16* arow = &As[wvM * 32 + mi * 16 + lr][0];
            v8bf alo = *(const v8bf*)(arow + klo);
            v8bf ahi = *(const v8bf*)(arow + klo + 8);
            v16bf afrag = __builtin_shufflevector(alo, ahi,
                0,1,2,3,4,5,6,7,8,9,10,11,12,13,14,15);
            #pragma unroll
            for (int ni = 0; ni < 2; ++ni) {
                const __bf16* brow = &Bs[wvN * 32 + ni * 16 + lr][0];
                v8bf blo = *(const v8bf*)(brow + kbB);
                v8bf bhi = *(const v8bf*)(brow + kbB + 8);
                v16bf bfrag = __builtin_shufflevector(blo, bhi,
                    0,1,2,3,4,5,6,7,8,9,10,11,12,13,14,15);
                acc[mi][ni] = __builtin_amdgcn_wmma_f32_16x16x32_bf16(
                    false, afrag, false, bfrag, (short)0, acc[mi][ni], false, false);
            }
        }
    }

    // epilogue: C layout -> lane(0-15): M=r,N=lane ; lane(16-31): M=r+8,N=lane-16
    const int mOff = (lane & 16) ? 8 : 0;
    const int nLoc = lane & 15;
    #pragma unroll
    for (int mi = 0; mi < 2; ++mi) {
        #pragma unroll
        for (int ni = 0; ni < 2; ++ni) {
            const int mBase = m0 + wvM * 32 + mi * 16 + mOff;
            const int nIdx  = n0 + wvN * 32 + ni * 16 + nLoc;
            #pragma unroll
            for (int r = 0; r < 8; ++r)
                qkv[(size_t)(mBase + r) * 3072 + nIdx] = f2bf_u(acc[mi][ni][r]);
        }
    }
}

// ============================================================
// Kernel 2: sliding-window attention, one wave per (b, t, h)
// ============================================================
__global__ __launch_bounds__(256)
void k_attn(const unsigned short* __restrict__ qkv, unsigned short* __restrict__ msg)
{
    __shared__ unsigned short qs[8][HDIM];  // per-wave q staging
    const int lane = threadIdx.x & 31;
    const int wv   = threadIdx.x >> 5;
    const int task = blockIdx.x * 8 + wv;   // (m*NHEAD + h)
    const int h = task & (NHEAD - 1);
    const int m = task >> 4;
    const int t = m & (TLEN - 1);
    const int b = m >> 12;

    // stage q (64 bf16) into LDS: lane l holds dims {2l, 2l+1}
    const unsigned short* qrow = qkv + (size_t)m * 3072 + h * HDIM;
    *(unsigned int*)&qs[wv][2 * lane] = *(const unsigned int*)(qrow + 2 * lane);
    __syncthreads();

    // lanes 0..15: dot(q, k[t-1-lane]) -> tau
    float tau = 0.f;
    const int srcl = t - 1 - lane;
    if (lane < WIN && srcl >= 0) {
        const unsigned short* krow =
            qkv + (size_t)((b << 12) + srcl) * 3072 + DIMD + h * 128;
        float dot = 0.f;
        #pragma unroll 8
        for (int d = 0; d < HDIM; d += 2) {
            unsigned int kk = *(const unsigned int*)(krow + d);
            unsigned int qq = *(const unsigned int*)&qs[wv][d];
            dot += bf2f((unsigned short)qq) * bf2f((unsigned short)kk);
            dot += bf2f((unsigned short)(qq >> 16)) * bf2f((unsigned short)(kk >> 16));
        }
        tau = 1.f / (1.f + __expf(-dot * 0.125f));   // SCALE = sqrt(64) = 8
    }

    // msg[d] = sum_w tau_w * v_w[d]; lane owns dims {2l, 2l+1}
    float acc0 = 0.f, acc1 = 0.f;
    #pragma unroll
    for (int w = 0; w < WIN; ++w) {
        const float tw = __shfl(tau, w, 32);
        const int src = t - 1 - w;
        if (src >= 0) {
            const unsigned short* vrow =
                qkv + (size_t)((b << 12) + src) * 3072 + DIMD + h * 128 + HDIM;
            unsigned int vv = *(const unsigned int*)(vrow + 2 * lane);
            acc0 += tw * bf2f((unsigned short)vv);
            acc1 += tw * bf2f((unsigned short)(vv >> 16));
        }
    }
    unsigned int packed = (unsigned int)f2bf_u(acc0) | ((unsigned int)f2bf_u(acc1) << 16);
    *(unsigned int*)(msg + (size_t)m * DIMD + h * HDIM + 2 * lane) = packed;
}

// ============================================================
// Kernel 3: y = x + msg @ wo^T   (M=8192, N=1024, K=1024)
// ============================================================
__global__ __launch_bounds__(256)
void k_gemm_out(const unsigned short* __restrict__ msg, const float* __restrict__ wo,
                const float* __restrict__ x, float* __restrict__ y)
{
    __shared__ __bf16 As[128][32];
    __shared__ __bf16 Bs[64][32];

    const int tid  = threadIdx.x;
    const int lane = tid & 31;
    const int wv   = tid >> 5;
    const int wvM  = wv >> 1;
    const int wvN  = wv & 1;
    const int m0   = blockIdx.y * 128;
    const int n0   = blockIdx.x * 64;

    v8f acc[2][2];
    #pragma unroll
    for (int i = 0; i < 2; ++i)
        #pragma unroll
        for (int j = 0; j < 2; ++j)
            acc[i][j] = (v8f){0.f,0.f,0.f,0.f,0.f,0.f,0.f,0.f};

    const int aRow = tid >> 1;
    const int aCol = (tid & 1) * 16;
    const int bRow = tid >> 2;
    const int bCol = (tid & 3) * 8;

    const float* wrow = wo + (size_t)(n0 + bRow) * DIMD;
    const unsigned short* arowp = msg + (size_t)(m0 + aRow) * DIMD + aCol;

    const int klo = (lane & 16) ? 16 : 0;
    const int kbB = (lane & 16) ? 16 : 0;
    const int lr  = lane & 15;

    for (int kk = 0; kk < DIMD; kk += 32) {
        __syncthreads();
        // A tile already bf16: straight 16B copies
        *(uint4*)&As[aRow][aCol]     = *(const uint4*)(arowp + kk);
        *(uint4*)&As[aRow][aCol + 8] = *(const uint4*)(arowp + kk + 8);
        // B tile fp32 -> bf16
        const float* bp = wrow + kk + bCol;
        float4 b0 = *(const float4*)(bp + 0);
        float4 b1 = *(const float4*)(bp + 4);
        uint4 pb;
        pb.x = pack2(b0.x, b0.y); pb.y = pack2(b0.z, b0.w);
        pb.z = pack2(b1.x, b1.y); pb.w = pack2(b1.z, b1.w);
        *(uint4*)&Bs[bRow][bCol] = pb;
        __syncthreads();

        #pragma unroll
        for (int mi = 0; mi < 2; ++mi) {
            const __bf16* arow = &As[wvM * 32 + mi * 16 + lr][0];
            v8bf alo = *(const v8bf*)(arow + klo);
            v8bf ahi = *(const v8bf*)(arow + klo + 8);
            v16bf afrag = __builtin_shufflevector(alo, ahi,
                0,1,2,3,4,5,6,7,8,9,10,11,12,13,14,15);
            #pragma unroll
            for (int ni = 0; ni < 2; ++ni) {
                const __bf16* brow = &Bs[wvN * 32 + ni * 16 + lr][0];
                v8bf blo = *(const v8bf*)(brow + kbB);
                v8bf bhi = *(const v8bf*)(brow + kbB + 8);
                v16bf bfrag = __builtin_shufflevector(blo, bhi,
                    0,1,2,3,4,5,6,7,8,9,10,11,12,13,14,15);
                acc[mi][ni] = __builtin_amdgcn_wmma_f32_16x16x32_bf16(
                    false, afrag, false, bfrag, (short)0, acc[mi][ni], false, false);
            }
        }
    }

    const int mOff = (lane & 16) ? 8 : 0;
    const int nLoc = lane & 15;
    #pragma unroll
    for (int mi = 0; mi < 2; ++mi) {
        #pragma unroll
        for (int ni = 0; ni < 2; ++ni) {
            const int mBase = m0 + wvM * 32 + mi * 16 + mOff;
            const int nIdx  = n0 + wvN * 32 + ni * 16 + nLoc;
            #pragma unroll
            for (int r = 0; r < 8; ++r) {
                const size_t off = (size_t)(mBase + r) * DIMD + nIdx;
                y[off] = x[off] + acc[mi][ni][r];
            }
        }
    }
}

// ============================================================
// Kernel 4: LayerNorm over last dim (1024), one block per row
// ============================================================
__global__ __launch_bounds__(256)
void k_ln(const float* __restrict__ y, const float* __restrict__ gamma,
          const float* __restrict__ beta, float* __restrict__ out)
{
    __shared__ float2 red[256];
    const int row = blockIdx.x;
    const int tid = threadIdx.x;
    const float* yr = y + (size_t)row * DIMD;
    const int c = tid * 4;
    float4 v = *(const float4*)(yr + c);
    float s  = v.x + v.y + v.z + v.w;
    float ss = v.x * v.x + v.y * v.y + v.z * v.z + v.w * v.w;
    red[tid] = make_float2(s, ss);
    __syncthreads();
    #pragma unroll
    for (int off = 128; off > 0; off >>= 1) {
        if (tid < off) {
            red[tid].x += red[tid + off].x;
            red[tid].y += red[tid + off].y;
        }
        __syncthreads();
    }
    const float mu  = red[0].x * (1.f / DIMD);
    const float var = red[0].y * (1.f / DIMD) - mu * mu;
    const float inv = rsqrtf(var + 1e-5f);
    float4 g  = *(const float4*)(gamma + c);
    float4 bt = *(const float4*)(beta + c);
    float4 o;
    o.x = (v.x - mu) * inv * g.x + bt.x;
    o.y = (v.y - mu) * inv * g.y + bt.y;
    o.z = (v.z - mu) * inv * g.z + bt.z;
    o.w = (v.w - mu) * inv * g.w + bt.w;
    *(float4*)(out + (size_t)row * DIMD + c) = o;
}

// ============================================================
extern "C" void kernel_launch(void* const* d_in, const int* in_sizes, int n_in,
                              void* d_out, int out_size, void* d_ws, size_t ws_size,
                              hipStream_t stream) {
    const float* x     = (const float*)d_in[0];
    const float* wq    = (const float*)d_in[1];
    const float* wkv   = (const float*)d_in[2];
    const float* wo    = (const float*)d_in[3];
    const float* gamma = (const float*)d_in[4];
    const float* beta  = (const float*)d_in[5];
    float* out = (float*)d_out;

    // workspace layout (96 MB total):
    //   qkv bf16 [8192][3072]  (48 MB)
    //   msg bf16 [8192][1024]  (16 MB)
    //   y   f32  [8192][1024]  (32 MB)
    unsigned short* qkv = (unsigned short*)d_ws;
    unsigned short* msg = qkv + (size_t)MTOT * 3072;
    float* y = (float*)(msg + (size_t)MTOT * DIMD);

    k_gemm_qkv<<<dim3(3072 / 64, MTOT / 128), 256, 0, stream>>>(x, wq, wkv, qkv);
    k_attn<<<dim3(MTOT * NHEAD / 8), 256, 0, stream>>>(qkv, msg);
    k_gemm_out<<<dim3(DIMD / 64, MTOT / 128), 256, 0, stream>>>(msg, wo, x, y);
    k_ln<<<dim3(MTOT), 256, 0, stream>>>(y, gamma, beta, out);
}
